// Attention_30777735643372
// MI455X (gfx1250) — matrix operand order repaired and checked by reading, not compiled, not run on hardware
//
#include <hip/hip_runtime.h>
#include <hip/hip_bf16.h>

// ---------------------------------------------------------------------------
// Fused multi-head attention for MI455X (gfx1250, wave32, WMMA).
// B=2 S=2048 D=1024 N=16 H=64.  All GEMMs via v_wmma_f32_16x16x32_f16.
// GEMM staging uses CDNA5 async global->LDS DMA (ASYNCcnt) with LDS double
// buffering so the tensor pipe overlaps the DMA.
// Reference quirk: masked (future) scores are set to EPS=1e-10 (NOT -inf),
// so softmax runs over ALL keys with masked entries substituted by 1e-10.
// ---------------------------------------------------------------------------

typedef __attribute__((ext_vector_type(8)))  _Float16 v8h;
typedef __attribute__((ext_vector_type(16))) _Float16 v16h;
typedef __attribute__((ext_vector_type(8)))  float    v8f;

#define B_   2
#define S_   2048
#define D_   1024
#define N_   16
#define H_   64
#define NH_  1024
#define M_   (B_ * S_)          // 4096
#define EPS_ 1e-10f
#define SCALE_ 0.125f           // 1/sqrt(64)

static __device__ __forceinline__ v16h cat16(v8h lo, v8h hi) {
  return __builtin_shufflevector(lo, hi, 0, 1, 2, 3, 4, 5, 6, 7,
                                         8, 9, 10, 11, 12, 13, 14, 15);
}

// A fragment (16x32 f16, M x K). Source: row-major [m][k] halves, row stride
// `strideH`. ISA layout: lane<16 -> kb=0, lane>=16 -> kb=8;
// elements 0..7 = K kb..kb+7, elements 8..15 = K 16+kb..16+kb+7.
static __device__ __forceinline__ v16h load_a_frag(const _Float16* base,
                                                   int strideH, int lane) {
  const int m  = lane & 15;
  const int kb = (lane >> 4) << 3;            // 0 or 8
  const _Float16* p = base + (size_t)m * strideH + kb;
  return cat16(*(const v8h*)p, *(const v8h*)(p + 16));
}

// B fragment (32x16 f16, K x N) loaded from B^T storage: row-major [n][k]
// halves, row stride `strideH`. ISA layout: lane<16 -> kb=0, lane>=16 -> kb=8;
// elements 0..7 = K kb..kb+7, elements 8..15 = K 16+kb..16+kb+7, column N = lane&15.
static __device__ __forceinline__ v16h load_b_frag(const _Float16* base,
                                                   int strideH, int lane) {
  const int n  = lane & 15;
  const int kb = (lane >> 4) << 3;            // 0 or 8
  const _Float16* p = base + (size_t)n * strideH + kb;
  return cat16(*(const v8h*)p, *(const v8h*)(p + 16));
}

static __device__ __forceinline__ v8f wmma_f16(v16h a, v16h b, v8f c) {
  return __builtin_amdgcn_wmma_f32_16x16x32_f16(
      /*neg_a=*/false, a, /*neg_b=*/false, b,
      /*c_mod=*/(short)0, c, /*reuse_a=*/false, /*reuse_b=*/false);
}

// CDNA5 async global->LDS 16B copy (VGPR-bypassing DMA, tracked by ASYNCcnt).
// LDS byte offset = low 32 bits of the flat shared-memory address (ISA 10.2:
// LDS aperture truncates flat addresses to the wave-relative LDS offset).
static __device__ __forceinline__ void async_ld_b128(void* lds,
                                                     const void* gaddr) {
  const unsigned l = (unsigned)(uintptr_t)lds;
  const unsigned long long g = (unsigned long long)(uintptr_t)gaddr;
  asm volatile("global_load_async_to_lds_b128 %0, %1, off"
               :
               : "v"(l), "v"(g)
               : "memory");
}
static __device__ __forceinline__ void wait_async0() {
  asm volatile("s_wait_asynccnt 0x0" ::: "memory");
}

// ---------------------------------------------------------------------------
// Kernel 1: convert residual to f16; build transposed f16 weight layouts.
//   Xh  [M][D]        = (f16) residual
//   WxT [col][d]      = Wx[n][d][h],  col = n*64+h   (x in {q,k,v})
//   WoT [e][col]      = Wo[n][h][e]                  ([D][NH])
// ---------------------------------------------------------------------------
__global__ __launch_bounds__(256) void prep_kernel(
    const float* __restrict__ res,
    const float* __restrict__ Wq, const float* __restrict__ Wk,
    const float* __restrict__ Wv, const float* __restrict__ Wo,
    _Float16* __restrict__ Xh,
    _Float16* __restrict__ WqT, _Float16* __restrict__ WkT,
    _Float16* __restrict__ WvT, _Float16* __restrict__ WoT) {
  const int i = blockIdx.x * 256 + threadIdx.x;        // 0 .. 4M-1
  Xh[i] = (_Float16)res[i];
  if (i < (1 << 20)) {
    const int col = i >> 10;                           // also 'e' for WoT
    const int d   = i & 1023;                          // also 'col' for WoT
    const int n   = col >> 6, h = col & 63;
    const int src = n * (D_ * H_) + d * H_ + h;
    WqT[i] = (_Float16)Wq[src];
    WkT[i] = (_Float16)Wk[src];
    WvT[i] = (_Float16)Wv[src];
    WoT[i] = (_Float16)Wo[(size_t)d * D_ + col];       // Wo[col2*D + e]
  }
}

// ---------------------------------------------------------------------------
// Shared GEMM tile machinery: block = 256 thr = 8 waves; tile M=64 x N=128;
// each wave computes a 16x64 strip via 4 C fragments.  LDS is double buffered;
// tiles are staged with async global->LDS b128 DMA issued one K-step ahead.
// ---------------------------------------------------------------------------
struct GemmTiles {
  _Float16 As[2][64 * 32];
  _Float16 Bs[2][128 * 32];
};

static __device__ __forceinline__ void stage_tile_async(
    GemmTiles& t, int buf, const _Float16* __restrict__ Arow,
    const _Float16* __restrict__ Brow, int k0, int tid) {
  // A: 64 rows x 32 halves; one 16B async DMA per thread.
  const int r = tid >> 2, s8 = (tid & 3) << 3;
  async_ld_b128(&t.As[buf][r * 32 + s8], Arow + (size_t)r * D_ + k0 + s8);
  // B^T: 128 rows x 32 halves; two 16B async DMAs per thread.
  const int rb = tid >> 1, sb = (tid & 1) << 4;
  const _Float16* src = Brow + (size_t)rb * D_ + k0 + sb;
  async_ld_b128(&t.Bs[buf][rb * 32 + sb], src);
  async_ld_b128(&t.Bs[buf][rb * 32 + sb + 8], src + 8);
}

// ---------------------------------------------------------------------------
// Kernel 2: QKV projection GEMM.  C[m][col] = Xh[m][:] . WT[col][:] + bias.
// z = blockIdx.z selects {Q,K,V}.  Q,K stored [b][n][s][h]; V stored
// transposed [b][n][h][s] so flash P.V B-fragments are contiguous.
// ---------------------------------------------------------------------------
__global__ __launch_bounds__(256) void qkv_kernel(
    const _Float16* __restrict__ Xh, const _Float16* __restrict__ WT0,
    const float* __restrict__ bq, const float* __restrict__ bk,
    const float* __restrict__ bv, _Float16* __restrict__ dst0) {
  __shared__ GemmTiles t;
  const int tid  = threadIdx.x;
  const int lane = tid & 31;
  const int wave = tid >> 5;
  const int wm   = wave & 3;            // M sub-tile (16 rows)
  const int wn   = wave >> 2;           // 0/1 -> N half (64 cols)
  const int z    = blockIdx.z;
  const int m0   = blockIdx.x * 64;
  const int n0   = blockIdx.y * 128;

  const _Float16* WT   = WT0 + (size_t)z * (1u << 20);
  const float*    bias = (z == 0) ? bq : (z == 1) ? bk : bv;
  _Float16*       dst  = dst0 + (size_t)z * (size_t)M_ * NH_;

  const _Float16* Arow = Xh + (size_t)m0 * D_;
  const _Float16* Brow = WT + (size_t)n0 * D_;

  v8f acc[4] = {};

  stage_tile_async(t, 0, Arow, Brow, 0, tid);
  wait_async0();
  __syncthreads();

  for (int k0 = 0; k0 < D_; k0 += 32) {
    const int cur = (k0 >> 5) & 1;
    if (k0 + 32 < D_)
      stage_tile_async(t, cur ^ 1, Arow, Brow, k0 + 32, tid);

    const v16h a = load_a_frag(t.As[cur] + wm * 16 * 32, 32, lane);
    for (int f = 0; f < 4; ++f) {
      const v16h b = load_b_frag(t.Bs[cur] + (wn * 64 + f * 16) * 32, 32, lane);
      acc[f] = wmma_f16(a, b, acc[f]);
    }
    wait_async0();          // next tile landed in LDS
    __syncthreads();        // everyone done reading `cur` (dscnt flushed)
  }

  // Epilogue: bias + scatter (Q/K row-major per head; V transposed).
  const int rhi = (lane >> 4) << 3;
  for (int f = 0; f < 4; ++f) {
    const int col = n0 + wn * 64 + f * 16 + (lane & 15);
    const float bc = bias[col];                  // [N,H] flat == col
    const int n = col >> 6, h = col & 63;
    for (int r = 0; r < 8; ++r) {
      const int m = m0 + wm * 16 + r + rhi;
      const int b = m >> 11, s = m & (S_ - 1);
      const float v = acc[f][r] + bc;
      size_t o;
      if (z == 2) o = ((size_t)(b * N_ + n) * H_ + h) * S_ + s;   // V^T
      else        o = ((size_t)(b * N_ + n) * S_ + s) * H_ + h;   // Q,K
      dst[o] = (_Float16)v;
    }
  }
}

// ---------------------------------------------------------------------------
// Kernel 3: flash attention with the EPS-mask quirk (full-row softmax).
// Block = 8 waves; each wave owns a 16-query x 64-h output tile; block covers
// 128 queries of one (b,n).  Per 32-key chunk: 4 score WMMAs, online softmax
// (shfl_xor butterflies within the 16-lane groups of the C layout), P staged
// through LDS (C-layout -> A-layout), 4 P.V WMMAs against V^T.
// ---------------------------------------------------------------------------
__global__ __launch_bounds__(256) void flash_kernel(
    const _Float16* __restrict__ Qh, const _Float16* __restrict__ Kh,
    const _Float16* __restrict__ Vt, _Float16* __restrict__ Wd) {
  __shared__ _Float16 pbuf[8 * 16 * 32];
  const int tid = threadIdx.x, lane = tid & 31, wave = tid >> 5;
  const int bn = blockIdx.y;                       // b*N + n
  const int q0 = blockIdx.x * 128 + wave * 16;

  const _Float16* qp = Qh + (size_t)bn * S_ * H_;  // [s][h]
  const _Float16* kp = Kh + (size_t)bn * S_ * H_;  // [s][h]
  const _Float16* vp = Vt + (size_t)bn * H_ * S_;  // [h][s]
  _Float16* pb = pbuf + wave * (16 * 32);

  // Q fragments resident for the whole key loop (K-dim = h, split 0..31/32..63)
  const v16h qa0 = load_a_frag(qp + (size_t)q0 * H_,      H_, lane);
  const v16h qa1 = load_a_frag(qp + (size_t)q0 * H_ + 32, H_, lane);

  v8f o[4] = {};
  float mrow[8], lrow[8];
  for (int r = 0; r < 8; ++r) { mrow[r] = -1e30f; lrow[r] = 0.f; }

  const int rhi = (lane >> 4) << 3;                // +8 for upper lane group

  for (int kc = 0; kc < S_; kc += 32) {
    // ---- scores for 16q x 32k ----
    v8f s0 = {}, s1 = {};
    {
      const v16h kb0 = load_b_frag(kp + (size_t)kc * H_,            H_, lane);
      const v16h kb1 = load_b_frag(kp + (size_t)kc * H_ + 32,       H_, lane);
      s0 = wmma_f16(qa0, kb0, s0);
      s0 = wmma_f16(qa1, kb1, s0);
      const v16h kb2 = load_b_frag(kp + (size_t)(kc + 16) * H_,      H_, lane);
      const v16h kb3 = load_b_frag(kp + (size_t)(kc + 16) * H_ + 32, H_, lane);
      s1 = wmma_f16(qa0, kb2, s1);
      s1 = wmma_f16(qa1, kb3, s1);
    }
    if (kc + 32 < S_) {                            // global_prefetch_b8
      __builtin_prefetch(kp + (size_t)(kc + 32) * H_, 0, 1);
      __builtin_prefetch(vp + (size_t)(kc + 32), 0, 1);
    }

    // ---- scale, EPS-mask, online softmax, stage P ----
    const int kg0 = kc + (lane & 15);
    const int kg1 = kg0 + 16;
    for (int r = 0; r < 8; ++r) {
      const int qg = q0 + r + rhi;
      float a = s0[r] * SCALE_; if (qg < kg0) a = EPS_;   // quirk: eps, not -inf
      float b = s1[r] * SCALE_; if (qg < kg1) b = EPS_;
      float mx = fmaxf(a, b);
      for (int off = 1; off < 16; off <<= 1)
        mx = fmaxf(mx, __shfl_xor(mx, off, 32));          // within 16-lane group
      const float mnew = fmaxf(mrow[r], mx);
      const float corr = __expf(mrow[r] - mnew);
      const float e0 = __expf(a - mnew);
      const float e1 = __expf(b - mnew);
      float sum = e0 + e1;
      for (int off = 1; off < 16; off <<= 1)
        sum += __shfl_xor(sum, off, 32);
      lrow[r] = lrow[r] * corr + sum;
      mrow[r] = mnew;
      for (int f = 0; f < 4; ++f) o[f][r] *= corr;        // rescale O row
      const int ql = r + rhi;
      pb[ql * 32 + (lane & 15)]      = (_Float16)e0;      // C-layout -> LDS
      pb[ql * 32 + (lane & 15) + 16] = (_Float16)e1;
    }

    // ---- P.V : A = P (16x32 from LDS), B^T = Vt rows ----
    const v16h pa = load_a_frag(pb, 32, lane);
    for (int f = 0; f < 4; ++f) {
      const v16h vb = load_b_frag(vp + (size_t)(f * 16) * S_ + kc, S_, lane);
      o[f] = wmma_f16(pa, vb, o[f]);
    }
  }

  // ---- normalize and store weighted [m][n*64+h] f16 ----
  const int bi = bn >> 4, ni = bn & 15;
  for (int f = 0; f < 4; ++f) {
    const int col = ni * 64 + f * 16 + (lane & 15);
    for (int r = 0; r < 8; ++r) {
      const int qg = q0 + r + rhi;
      const size_t m = (size_t)bi * S_ + qg;
      Wd[m * NH_ + col] = (_Float16)(o[f][r] / lrow[r]);
    }
  }
}

// ---------------------------------------------------------------------------
// Kernel 4: output projection.  out[m][e] = Wd[m][:] . WoT[e][:] + Bo[e].
// Same async double-buffered tiling as qkv_kernel; fp32 store to d_out.
// ---------------------------------------------------------------------------
__global__ __launch_bounds__(256) void out_kernel(
    const _Float16* __restrict__ Wd, const _Float16* __restrict__ WoT,
    const float* __restrict__ bo, float* __restrict__ out) {
  __shared__ GemmTiles t;
  const int tid  = threadIdx.x;
  const int lane = tid & 31;
  const int wave = tid >> 5;
  const int wm   = wave & 3;
  const int wn   = wave >> 2;
  const int m0   = blockIdx.x * 64;
  const int n0   = blockIdx.y * 128;

  const _Float16* Arow = Wd  + (size_t)m0 * NH_;
  const _Float16* Brow = WoT + (size_t)n0 * NH_;

  v8f acc[4] = {};

  stage_tile_async(t, 0, Arow, Brow, 0, tid);
  wait_async0();
  __syncthreads();

  for (int k0 = 0; k0 < NH_; k0 += 32) {
    const int cur = (k0 >> 5) & 1;
    if (k0 + 32 < NH_)
      stage_tile_async(t, cur ^ 1, Arow, Brow, k0 + 32, tid);

    const v16h a = load_a_frag(t.As[cur] + wm * 16 * 32, 32, lane);
    for (int f = 0; f < 4; ++f) {
      const v16h b = load_b_frag(t.Bs[cur] + (wn * 64 + f * 16) * 32, 32, lane);
      acc[f] = wmma_f16(a, b, acc[f]);
    }
    wait_async0();
    __syncthreads();
  }

  const int rhi = (lane >> 4) << 3;
  for (int f = 0; f < 4; ++f) {
    const int e  = n0 + wn * 64 + f * 16 + (lane & 15);
    const float bc = bo[e];
    for (int r = 0; r < 8; ++r) {
      const int m = m0 + wm * 16 + r + rhi;
      out[(size_t)m * D_ + e] = acc[f][r] + bc;
    }
  }
}

// ---------------------------------------------------------------------------
// Workspace layout (f16 halves):
//   Xh   4Mi | WqT 1Mi | WkT 1Mi | WvT 1Mi | WoT 1Mi |
//   Qh   4Mi | Kh  4Mi | Vt  4Mi | Wd  4Mi            -> 24Mi halves = 48 MiB
// ---------------------------------------------------------------------------
extern "C" void kernel_launch(void* const* d_in, const int* in_sizes, int n_in,
                              void* d_out, int out_size, void* d_ws,
                              size_t ws_size, hipStream_t stream) {
  (void)in_sizes; (void)n_in; (void)out_size; (void)ws_size;
  const float* res = (const float*)d_in[0];
  const float* Wk  = (const float*)d_in[1];
  const float* Wq  = (const float*)d_in[2];
  const float* Wv  = (const float*)d_in[3];
  const float* Wo  = (const float*)d_in[4];
  const float* Bk  = (const float*)d_in[5];
  const float* Bq  = (const float*)d_in[6];
  const float* Bv  = (const float*)d_in[7];
  const float* Bo  = (const float*)d_in[8];

  _Float16* ws  = (_Float16*)d_ws;
  _Float16* Xh  = ws;
  _Float16* WqT = Xh  + (size_t)(4u << 20);
  _Float16* WkT = WqT + (size_t)(1u << 20);
  _Float16* WvT = WkT + (size_t)(1u << 20);
  _Float16* WoT = WvT + (size_t)(1u << 20);
  _Float16* Qh  = WoT + (size_t)(1u << 20);
  _Float16* Kh  = Qh  + (size_t)(4u << 20);
  _Float16* Vt  = Kh  + (size_t)(4u << 20);
  _Float16* Wd  = Vt  + (size_t)(4u << 20);

  prep_kernel<<<(4u << 20) / 256, 256, 0, stream>>>(
      res, Wq, Wk, Wv, Wo, Xh, WqT, WkT, WvT, WoT);
  qkv_kernel<<<dim3(M_ / 64, NH_ / 128, 3), 256, 0, stream>>>(
      Xh, WqT, Bq, Bk, Bv, Qh);
  flash_kernel<<<dim3(S_ / 128, B_ * N_), 256, 0, stream>>>(Qh, Kh, Vt, Wd);
  out_kernel<<<dim3(M_ / 64, D_ / 128), 256, 0, stream>>>(
      Wd, WoT, Bo, (float*)d_out);
}
